// GlobalPointer_18184891531945
// MI455X (gfx1250) — matrix operand order repaired and checked by reading, not compiled, not run on hardware
//
#include <hip/hip_runtime.h>
#include <hip/hip_bf16.h>

// ---------------------------------------------------------------------------
// GlobalPointer head for MI455X (gfx1250, wave32).
//   out[b,h,m,n] = mask( RoPE(x@W+b).q[b,m,h,:] . RoPE(x@W+b).k[b,n,h,:] ) / 8
// Stage 0: transpose W (768x1152) -> Wt (1152x768)      [one-time, LDS tiled]
// Stage 1: projection GEMM via v_wmma_f32_16x16x4_f32 (full f32, K=768),
//          bias + RoPE epilogue, store q/k as f16 to workspace.
// Stage 2: logits GEMM via v_wmma_f32_16x16x32_f16 (f32 accum, K=64),
//          padding + causal mask + 1/sqrt(64) scale, f32 store.
// Causal tiles strictly below the diagonal skip compute: acc is numerically
// annihilated by the -1e12 subtraction in f32 (ulp(1e12)=2^16).
// ---------------------------------------------------------------------------

#define HIDDEN 768
#define NOUT   1152   // ENT(9) * INNER(64) * 2
#define SEQ    1024
#define BATCH  8
#define ENT    9
#define INNER  64

typedef __attribute__((ext_vector_type(2)))  float    v2f;
typedef __attribute__((ext_vector_type(8)))  float    v8f;
typedef __attribute__((ext_vector_type(8)))  _Float16 v8h;
typedef __attribute__((ext_vector_type(16))) _Float16 v16h;

// ---------------------------------------------------------------------------
// Kernel 0: W (K=768 x N=1152) -> Wt (N x K), LDS-tiled 16x16 transpose.
// ---------------------------------------------------------------------------
__global__ void wtrans_kernel(const float* __restrict__ W, float* __restrict__ Wt) {
  __shared__ float tile[16][17];
  const int nb = blockIdx.x * 16;          // 0..1151
  const int kb = blockIdx.y * 16;          // 0..767
  const int tx = threadIdx.x & 15;
  const int ty = threadIdx.x >> 4;         // 256 threads = 16x16
  tile[ty][tx] = W[(size_t)(kb + ty) * NOUT + (nb + tx)];
  __syncthreads();
  Wt[(size_t)(nb + ty) * HIDDEN + (kb + tx)] = tile[tx][ty];
}

// ---------------------------------------------------------------------------
// Kernel 1: projection GEMM (M=8192, N=1152, K=768) + bias + RoPE -> f16 qk.
// One wave computes a 16(M) x 64(N) strip = 4 C fragments, reusing the A
// fragment across the 4 N-tiles. Block = 8 waves stacked along M.
// f32 WMMA fragment layout (ISA 7.12.2):
//   A 16x4 : lane L row m=L%16; VGPR0/1 = K {2*(L/16), 2*(L/16)+1}
//   B 4x16 : lane L col n=L%16; VGPR0/1 = K {2*(L/16), 2*(L/16)+1}
//   C 16x16: lane L col n=L%16; VGPR r  = row r + 8*(L/16)
// ---------------------------------------------------------------------------
__global__ void proj_rope_kernel(const float* __restrict__ x,
                                 const float* __restrict__ Wt,
                                 const float* __restrict__ bias,
                                 _Float16* __restrict__ qk) {
  const int lane  = threadIdx.x & 31;
  const int wave  = threadIdx.x >> 5;
  const int mtile = blockIdx.y * 8 + wave;   // 0..511  (16 rows each)
  const int n64   = blockIdx.x;              // 0..17   (64 cols each)
  const int khalf = lane >> 4;               // 0 | 1
  const int nl    = lane & 15;

  const int  arow = mtile * 16 + nl;                       // A row this lane feeds
  const float* __restrict__ xrow = x + (size_t)arow * HIDDEN;
  const int  ncol0 = n64 * 64 + nl;

  v8f acc[4] = {v8f{}, v8f{}, v8f{}, v8f{}};

  for (int k = 0; k < HIDDEN; k += 4) {
    const int kk = k + khalf * 2;
    const v2f a = *(const v2f*)(xrow + kk);
#pragma unroll
    for (int nt = 0; nt < 4; ++nt) {
      const int n = ncol0 + nt * 16;
      const v2f bfrag = *(const v2f*)(Wt + (size_t)n * HIDDEN + kk);
      acc[nt] = __builtin_amdgcn_wmma_f32_16x16x4_f32(
          false, a, false, bfrag, (short)0, acc[nt], false, false);
    }
  }

  // Epilogue: bias, RoPE (rotate-half partner = adjacent lane), f16 store.
  const float LOG1E4_OVER_DIM = 9.2103403719761836f / 64.0f; // ln(10000)/64
#pragma unroll
  for (int nt = 0; nt < 4; ++nt) {
    const int   n   = ncol0 + nt * 16;
    const float bn  = bias[n];
    const int   d   = n & 63;                         // RoPE dim index (q or k half)
    const float inv = __expf(-(float)(d & ~1) * LOG1E4_OVER_DIM); // 10000^(-2i/64)
    const bool  odd = (n & 1) != 0;
#pragma unroll
    for (int r = 0; r < 8; ++r) {
      const int   me = mtile * 16 + khalf * 8 + r;    // C element row (b*S+s)
      const float v  = acc[nt][r] + bn;
      const float pa = __shfl_xor(v, 1, 32);          // paired column (n^1) value
      const float rot = odd ? pa : -pa;               // rotate_half
      const float ang = (float)(me & (SEQ - 1)) * inv;
      float sn, cs;
      __sincosf(ang, &sn, &cs);
      qk[(size_t)me * NOUT + n] = (_Float16)(v * cs + rot * sn);
    }
  }
}

// ---------------------------------------------------------------------------
// Kernel 2: logits[b,h] = Q(1024x64) @ K^T, masked, /8.
// One wave computes 16(M) x 64(N); A frag reused over 4 N-tiles; two
// v_wmma_f32_16x16x32_f16 per tile cover K=64. Block = 8 waves along M.
// f16 fragment k order used here (contiguous, same for A and B):
//   A 16x32: lane L row m=L%16; halves[j]=K(16*(L/16)+j)
//   B 32x16: lane L col n=L%16; halves[j]=K(16*(L/16)+j)  -> contiguous 32B load
// ---------------------------------------------------------------------------
__global__ void logits_kernel(const _Float16* __restrict__ qk,
                              const float* __restrict__ pmask,
                              float* __restrict__ out) {
  const int lane  = threadIdx.x & 31;
  const int wave  = threadIdx.x >> 5;
  const int bh    = blockIdx.z;             // 0..71
  const int bb    = bh / ENT;
  const int h     = bh % ENT;
  const int mtile = blockIdx.y * 8 + wave;  // 0..63
  const int n64   = blockIdx.x;             // 0..15
  const int khalf = lane >> 4;
  const int nl    = lane & 15;

  v8f acc[4] = {v8f{}, v8f{}, v8f{}, v8f{}};

  const int mlo = mtile * 16;               // wave-uniform skip test (EXEC stays all-1s)
  const int nhi = n64 * 64 + 63;
  if (mlo <= nhi) {                         // not strictly below the causal diagonal
    const int m = mlo + nl;
    const _Float16* __restrict__ qrow =
        qk + (size_t)(bb * SEQ + m) * NOUT + h * 2 * INNER;
    const v8h a0lo = *(const v8h*)(qrow + khalf * 16);
    const v8h a0hi = *(const v8h*)(qrow + 8 + khalf * 16);
    const v8h a1lo = *(const v8h*)(qrow + 32 + khalf * 16);
    const v8h a1hi = *(const v8h*)(qrow + 40 + khalf * 16);
    v16h A0, A1;
#pragma unroll
    for (int i = 0; i < 8; ++i) {
      A0[i] = a0lo[i]; A0[8 + i] = a0hi[i];
      A1[i] = a1lo[i]; A1[8 + i] = a1hi[i];
    }
#pragma unroll
    for (int nt = 0; nt < 4; ++nt) {
      const int n = n64 * 64 + nt * 16 + nl;
      const _Float16* __restrict__ krow =
          qk + (size_t)(bb * SEQ + n) * NOUT + h * 2 * INNER + INNER;
      const v16h B0 = *(const v16h*)(krow + khalf * 16);
      const v16h B1 = *(const v16h*)(krow + 32 + khalf * 16);
      acc[nt] = __builtin_amdgcn_wmma_f32_16x16x32_f16(
          false, A0, false, B0, (short)0, acc[nt], false, false);
      acc[nt] = __builtin_amdgcn_wmma_f32_16x16x32_f16(
          false, A1, false, B1, (short)0, acc[nt], false, false);
    }
  }

  // Epilogue: padding mask, causal mask, scale by 1/sqrt(INNER)=0.125.
  const float* __restrict__ pmb = pmask + bb * SEQ;
  float* __restrict__ obase = out + (size_t)bh * SEQ * SEQ;
#pragma unroll
  for (int nt = 0; nt < 4; ++nt) {
    const int   n  = n64 * 64 + nt * 16 + nl;
    const float pm = pmb[n];
    const float madd = -(1.0f - pm) * 1.0e12f;
#pragma unroll
    for (int r = 0; r < 8; ++r) {
      const int   me = mtile * 16 + khalf * 8 + r;
      const float v  = acc[nt][r] * pm + madd - ((me > n) ? 1.0e12f : 0.0f);
      obase[(size_t)me * SEQ + n] = v * 0.125f;
    }
  }
}

// ---------------------------------------------------------------------------
extern "C" void kernel_launch(void* const* d_in, const int* in_sizes, int n_in,
                              void* d_out, int out_size, void* d_ws, size_t ws_size,
                              hipStream_t stream) {
  (void)in_sizes; (void)n_in; (void)out_size; (void)ws_size;
  const float* x     = (const float*)d_in[0];  // (8,1024,768)
  const float* pmask = (const float*)d_in[1];  // (8,1024)
  const float* W     = (const float*)d_in[2];  // (768,1152)
  const float* bias  = (const float*)d_in[3];  // (1152,)
  float* out = (float*)d_out;                  // (8,9,1024,1024)

  // Workspace: [ Wt : 1152*768 f32 | qk : 8192*1152 f16 ]  (~22.4 MB total)
  float*    Wt = (float*)d_ws;
  _Float16* qk = (_Float16*)((char*)d_ws + (size_t)NOUT * HIDDEN * sizeof(float));

  wtrans_kernel<<<dim3(NOUT / 16, HIDDEN / 16), 256, 0, stream>>>(W, Wt);
  proj_rope_kernel<<<dim3(NOUT / 64, (BATCH * SEQ) / (16 * 8)), 256, 0, stream>>>(
      x, Wt, bias, qk);
  logits_kernel<<<dim3(SEQ / 64, SEQ / (16 * 8), BATCH * ENT), 256, 0, stream>>>(
      qk, pmask, out);
}
